// LGCN_80255758893342
// MI455X (gfx1250) — matrix operand order repaired and checked by reading, not compiled, not run on hardware
//
#include <hip/hip_runtime.h>
#include <hip/hip_bf16.h>

// ---------------------------------------------------------------------------
// Problem constants (from reference)
// ---------------------------------------------------------------------------
constexpr int NN   = 50000;        // nodes
constexpr int SS   = 5;            // slots
constexpr int DIN  = 512;
constexpr int HH   = 64;
constexpr int LL   = 6;
constexpr int EE   = 800000;
constexpr int NS   = NN * SS;      // 250000 rows
constexpr int ES   = EE * SS;      // 4,000,000 replicated edges
constexpr int NT   = NS / 16;      // 15625 row-tiles of 16

typedef __attribute__((ext_vector_type(16))) __bf16 v16bf;
typedef __attribute__((ext_vector_type(8)))  float  v8f;

// Native fire-and-forget f32 atomic add (no return -> STOREcnt, device scope
// so the RMW is performed at L2, coherent across all WGPs). Avoids any chance
// of a safe-FP CAS-loop expansion on the hot scatter path.
__device__ __forceinline__ void atomic_add_f32_dev(float* p, float v) {
    asm volatile("global_atomic_add_f32 %0, %1, off scope:SCOPE_DEV"
                 :
                 : "v"(p), "v"(v)
                 : "memory");
}

// ---------------------------------------------------------------------------
// Weight prep: f32 -> bf16, transposed so B-columns are contiguous
// ---------------------------------------------------------------------------
__global__ void prep_wemb(const float* __restrict__ W, __bf16* __restrict__ Wt) {
    int i = blockIdx.x * blockDim.x + threadIdx.x;      // 64*512
    if (i >= HH * DIN) return;
    int n = i / DIN, k = i % DIN;
    Wt[i] = (__bf16)W[k * HH + n];                      // Wt[n][k] = W[k][n]
}

__global__ void prep_wgcn(const float* __restrict__ W, __bf16* __restrict__ Wt) {
    int i = blockIdx.x * blockDim.x + threadIdx.x;      // 6*64*64
    if (i >= LL * HH * HH) return;
    int l = i / (HH * HH); int r = i % (HH * HH);
    int n = r / HH, k = r % HH;
    Wt[i] = (__bf16)W[l * HH * HH + k * HH + n];
}

// ---------------------------------------------------------------------------
// WMMA GEMM: D[NS x 64] = A[NS x K] (f32 -> bf16) @ Bt^T (Bt is [64][K] bf16)
// One wave computes a 16x64 tile: 4 accumulators of v8f.
// A-layout (16-bit A 16x32, contiguous k like B): lane<16 -> K 0-15; lane>=16 -> K 16-31.
// B-layout: lane 0-15 = column N, K k..k+15; lane 16-31 same N, K k+16..k+31.
//
// NT_A:  stream A with non-temporal loads (read-once 512 MB batched_data,
//        preserves L2 residency of the 64 MB h/hw buffers for the scatter).
// FUSE:  also emit agg = hw * dinv^2 + gcn_b from the accumulators (saves a
//        full 128 MB read+write pass per GCN layer).
// ---------------------------------------------------------------------------
template <int K, bool NT_A, bool FUSE>
__global__ __launch_bounds__(256) void gemm16(const float* __restrict__ A,
                                              const __bf16* __restrict__ Bt,
                                              const float* __restrict__ bias,
                                              float* __restrict__ D,
                                              const float* __restrict__ dinv,
                                              float* __restrict__ D2) {
    int wid  = (blockIdx.x * blockDim.x + threadIdx.x) >> 5;
    int lane = threadIdx.x & 31;
    if (wid >= NT) return;                      // wave-uniform: EXEC stays all-ones
    const int row0 = wid << 4;
    const int m    = lane & 15;
    const int half = lane >> 4;

    const float*  arow = A  + (size_t)(row0 + m) * K + half * 16;
    const __bf16* bcol = Bt + (size_t)m * K + half * 16;

    v8f acc[4] = {};

    for (int k = 0; k < K; k += 32) {
        // A tile: two contiguous 8-float reads per lane, cast to bf16
        v16bf a;
        const float* ap = arow + k;
#pragma unroll
        for (int i = 0; i < 8; ++i) {
            float lo, hi;
            if constexpr (NT_A) {
                lo = __builtin_nontemporal_load(ap + i);
                hi = __builtin_nontemporal_load(ap + 8 + i);
            } else {
                lo = ap[i];
                hi = ap[8 + i];
            }
            a[i]     = (__bf16)lo;
            a[i + 8] = (__bf16)hi;
        }
#pragma unroll
        for (int t = 0; t < 4; ++t) {
            v16bf bv = *(const v16bf*)(bcol + (size_t)(t * 16) * K + k);
            acc[t] = __builtin_amdgcn_wmma_f32_16x16x32_bf16(
                false, a, false, bv, (short)0, acc[t], false, false);
        }
    }

    // D layout: VGPR i, lanes 0-15 -> M=i, N=lane; lanes 16-31 -> M=i+8
    if constexpr (FUSE) {
        // D = hw; D2 = hw * dinv^2 + gcn_b (scatter later accumulates into D2)
        float sn[8];
#pragma unroll
        for (int i = 0; i < 8; ++i) {
            float d = dinv[row0 + i + 8 * half];
            sn[i] = d * d;
        }
#pragma unroll
        for (int t = 0; t < 4; ++t) {
            int col = t * 16 + m;
            float gbv = bias[col];
#pragma unroll
            for (int i = 0; i < 8; ++i) {
                size_t idx = (size_t)(row0 + i + 8 * half) * HH + col;
                float v = acc[t][i];
                D[idx]  = v;
                D2[idx] = v * sn[i] + gbv;
            }
        }
    } else {
#pragma unroll
        for (int t = 0; t < 4; ++t) {
            int col = t * 16 + m;
            float bb = bias ? bias[col] : 0.0f;
#pragma unroll
            for (int i = 0; i < 8; ++i) {
                D[(size_t)(row0 + i + 8 * half) * HH + col] = acc[t][i] + bb;
            }
        }
    }
}

// ---------------------------------------------------------------------------
// Degree / normalization
// ---------------------------------------------------------------------------
__global__ void fill_ones(float* __restrict__ p, int n) {
    int i = blockIdx.x * blockDim.x + threadIdx.x;
    if (i < n) p[i] = 1.0f;
}

__global__ void deg_count(const int* __restrict__ ei, float* __restrict__ deg) {
    int g = blockIdx.x * blockDim.x + threadIdx.x;      // ES threads
    if (g >= ES) return;
    int s = g / EE, e = g - s * EE;
    int rd = ei[EE + e] + s * NN;
    atomic_add_f32_dev(&deg[rd], 1.0f);
}

__global__ void dinv_k(const float* __restrict__ deg, float* __restrict__ dinv, int n) {
    int i = blockIdx.x * blockDim.x + threadIdx.x;
    if (i < n) dinv[i] = rsqrtf(deg[i]);
}

__global__ void norm_k(const int* __restrict__ ei, const float* __restrict__ dinv,
                       float* __restrict__ nrm) {
    int g = blockIdx.x * blockDim.x + threadIdx.x;      // ES threads
    if (g >= ES) return;
    int s = g / EE, e = g - s * EE;
    int rs = ei[e] + s * NN;
    int rd = ei[EE + e] + s * NN;
    nrm[g] = dinv[rs] * dinv[rd];
}

// 64 consecutive lanes handle one replicated edge -> coalesced gather + atomics.
// hw/agg are 64 MB each: L2-resident on MI455X (192 MB L2), so the random edge
// traffic is mostly L2 hits; native f32 atomics RMW at L2.
__global__ __launch_bounds__(256) void scatter_k(const float* __restrict__ hw,
                                                 const int* __restrict__ ei,
                                                 const float* __restrict__ nrm,
                                                 float* __restrict__ agg) {
    long long g  = (long long)blockIdx.x * blockDim.x + threadIdx.x;
    int       c  = (int)(g & 63);
    long long eg = g >> 6;
    if (eg >= (long long)ES) return;
    int s = (int)(eg / EE);
    int e = (int)(eg - (long long)s * EE);
    int rs = ei[e] + s * NN;
    int rd = ei[EE + e] + s * NN;
    float v = hw[(size_t)rs * HH + c] * nrm[eg];
    atomic_add_f32_dev(&agg[(size_t)rd * HH + c], v);
}

// ---------------------------------------------------------------------------
// LayerNorm + attention epilogue. One wave32 per node; lane owns channels
// (lane, lane+32). Butterfly reductions via __shfl_xor.
// ---------------------------------------------------------------------------
__device__ __forceinline__ float wave_sum(float v) {
#pragma unroll
    for (int m = 16; m >= 1; m >>= 1) v += __shfl_xor(v, m, 32);
    return v;
}

__global__ __launch_bounds__(256) void ln_attn(const float* __restrict__ h,
                                               const float* __restrict__ g,
                                               const float* __restrict__ b,
                                               const float* __restrict__ aw,
                                               const float* __restrict__ ab,
                                               float* __restrict__ out) {
    int wid  = (blockIdx.x * blockDim.x + threadIdx.x) >> 5;
    int lane = threadIdx.x & 31;
    if (wid >= NN) return;

    const float* base = h + (size_t)wid * SS * HH;
    float g0 = g[lane], g1 = g[lane + 32];
    float b0 = b[lane], b1 = b[lane + 32];

    float ln[SS][2];
#pragma unroll
    for (int s = 0; s < SS; ++s) {
        float v0 = base[s * HH + lane];
        float v1 = base[s * HH + 32 + lane];
        float mu = wave_sum(v0 + v1) * (1.0f / HH);
        float sq = wave_sum(v0 * v0 + v1 * v1) * (1.0f / HH);
        float var = sq - mu * mu;
        float r = rsqrtf(var + 1e-5f);
        ln[s][0] = (v0 - mu) * r * g0 + b0;
        ln[s][1] = (v1 - mu) * r * g1 + b1;
    }

    float a0 = aw[lane], a1 = aw[lane + 32];
    float a2 = aw[HH + lane], a3 = aw[HH + 32 + lane];
    float bias = ab[0];

    float logit[SS - 1];
#pragma unroll
    for (int s = 1; s < SS; ++s) {
        float p = ln[0][0] * a0 + ln[0][1] * a1 + ln[s][0] * a2 + ln[s][1] * a3;
        logit[s - 1] = wave_sum(p) + bias;
    }
    float mx = fmaxf(fmaxf(logit[0], logit[1]), fmaxf(logit[2], logit[3]));
    float ex[SS - 1], den = 0.0f;
#pragma unroll
    for (int i = 0; i < SS - 1; ++i) { ex[i] = expf(logit[i] - mx); den += ex[i]; }
    float inv = 1.0f / den;

    // out[0 : N*64] = node, then N*4*64 of neigh*att
    out[(size_t)wid * HH + lane]      = ln[0][0];
    out[(size_t)wid * HH + 32 + lane] = ln[0][1];
    float* o2 = out + (size_t)NN * HH;
#pragma unroll
    for (int s = 1; s < SS; ++s) {
        float att = ex[s - 1] * inv;
        size_t ro = ((size_t)wid * (SS - 1) + (s - 1)) * HH;
        o2[ro + lane]      = ln[s][0] * att;
        o2[ro + 32 + lane] = ln[s][1] * att;
    }
}

// ---------------------------------------------------------------------------
// Launch
// ---------------------------------------------------------------------------
extern "C" void kernel_launch(void* const* d_in, const int* in_sizes, int n_in,
                              void* d_out, int out_size, void* d_ws, size_t ws_size,
                              hipStream_t stream) {
    const float* bd   = (const float*)d_in[0];   // (N,S,512)
    const int*   ei   = (const int*)  d_in[1];   // (2,E)
    const float* Wemb = (const float*)d_in[2];   // (512,64)
    const float* bemb = (const float*)d_in[3];   // (64,)
    const float* gW   = (const float*)d_in[4];   // (6,64,64)
    const float* gb   = (const float*)d_in[5];   // (6,64)
    const float* lng  = (const float*)d_in[6];
    const float* lnb  = (const float*)d_in[7];
    const float* aW   = (const float*)d_in[8];   // (128,1)
    const float* abz  = (const float*)d_in[9];   // (1,)
    float* out = (float*)d_out;

    char* ws = (char*)d_ws;
    float*  h0   = (float*)(ws);                       //  64,000,000 B
    float*  h1   = (float*)(ws + 64000000ULL);         //  64,000,000 B
    float*  hw   = (float*)(ws + 128000000ULL);        //  64,000,000 B
    float*  deg  = (float*)(ws + 192000000ULL);        //   1,000,000 B
    float*  dinv = (float*)(ws + 193000000ULL);        //   1,000,000 B
    float*  nrm  = (float*)(ws + 194000000ULL);        //  16,000,000 B
    __bf16* wtE  = (__bf16*)(ws + 210000000ULL);       //      65,536 B
    __bf16* wtG  = (__bf16*)(ws + 210100000ULL);       //      49,152 B

    // weight prep
    prep_wemb<<<(HH * DIN + 255) / 256, 256, 0, stream>>>(Wemb, wtE);
    prep_wgcn<<<(LL * HH * HH + 255) / 256, 256, 0, stream>>>(gW, wtG);

    // degrees & norms
    fill_ones<<<(NS + 255) / 256, 256, 0, stream>>>(deg, NS);
    deg_count<<<ES / 256, 256, 0, stream>>>(ei, deg);
    dinv_k<<<(NS + 255) / 256, 256, 0, stream>>>(deg, dinv, NS);
    norm_k<<<ES / 256, 256, 0, stream>>>(ei, dinv, nrm);

    // embedding GEMM (K=512), A streamed non-temporally
    int gemmBlocks = (NT + 7) / 8;                 // 8 waves per 256-thread block
    gemm16<DIN, true, false><<<gemmBlocks, 256, 0, stream>>>(
        bd, wtE, bemb, h0, nullptr, nullptr);

    // GCN layers: GEMM writes hw and (fused) agg-init into nxt; scatter adds
    float* cur = h0;
    float* nxt = h1;
    for (int l = 0; l < LL; ++l) {
        gemm16<HH, false, true><<<gemmBlocks, 256, 0, stream>>>(
            cur, wtG + (size_t)l * HH * HH, gb + l * HH, hw, dinv, nxt);
        scatter_k<<<(long long)ES * HH / 256, 256, 0, stream>>>(hw, ei, nrm, nxt);
        float* t = cur; cur = nxt; nxt = t;
    }

    // LayerNorm + attention epilogue
    ln_attn<<<(NN * 32) / 256, 256, 0, stream>>>(cur, lng, lnb, aW, abz, out);
}
